// Encoder_17995912970597
// MI455X (gfx1250) — matrix operand order repaired and checked by reading, not compiled, not run on hardware
//
#include <hip/hip_runtime.h>

// ---------------- problem constants ----------------
#define B_      8
#define N_      8192
#define M_      64
#define KNN_    3
#define SOMK_   9
#define KN_     (KNN_*N_)        // 24576 rows per batch in stage 1
#define ROWS1_  (B_*KN_)         // 196608
#define NCHUNK_ 8
#define CHR_    (ROWS1_/NCHUNK_) // 24576 rows / chunk (divisible by 128)
#define CF_     384              // node feature width
#define INV_BN_ 0.9999950000374997f

typedef unsigned short u16;
typedef unsigned int   u32;
typedef __attribute__((ext_vector_type(16))) __bf16 v16bf;
typedef __attribute__((ext_vector_type(8)))  float  v8f;
typedef int v4i_ __attribute__((vector_size(16)));   // matches builtin param type

union FragB { uint4 q[2]; v16bf v; };

__device__ __forceinline__ u16 f2bf(float f) {
    u32 u = __float_as_uint(f);
    u32 r = (u + 0x7FFFu + ((u >> 16) & 1u)) >> 16;   // RNE
    return (u16)r;
}
__device__ __forceinline__ float bf2f(u16 h) {
    return __uint_as_float(((u32)h) << 16);
}

// ---------------- CDNA5 async global->LDS (ASYNCcnt-tracked) ----------------
#if defined(__has_builtin)
#  if __has_builtin(__builtin_amdgcn_global_load_async_to_lds_b128)
#    define HAVE_ASYNC_B128 1
#  endif
#  if __has_builtin(__builtin_amdgcn_s_wait_asynccnt)
#    define HAVE_WAIT_ASYNC 1
#  endif
#endif

__device__ __forceinline__ void async_cp16(const u16* src, u16* dst) {
#ifdef HAVE_ASYNC_B128
    __builtin_amdgcn_global_load_async_to_lds_b128(
        (__attribute__((address_space(1))) v4i_*)src,
        (__attribute__((address_space(3))) v4i_*)dst, 0, 0);
#else
    u32 ldsa = (u32)(size_t)dst;                      // low 32 bits = LDS offset
    asm volatile("global_load_async_to_lds_b128 %0, %1, off"
                 :: "v"(ldsa), "v"((const void*)src) : "memory");
#endif
}

#ifdef HAVE_WAIT_ASYNC
#  define WAIT_ASYNC(n) __builtin_amdgcn_s_wait_asynccnt(n)
#else
#  define WAIT_ASYNC(n) asm volatile("s_wait_asynccnt " #n ::: "memory")
#endif

// ---------------- K1: top-3 SOM assignment per point ----------------
__global__ __launch_bounds__(256) void k_assign(const float* __restrict__ x,
                                                const float* __restrict__ node,
                                                int* __restrict__ minidx) {
    int gid = blockIdx.x * 256 + threadIdx.x;     // B*N threads
    int b = gid >> 13;                            // /8192
    int n = gid & (N_ - 1);
    __shared__ float nd[192];
    if (threadIdx.x < 192) nd[threadIdx.x] = node[b * 192 + threadIdx.x];
    __syncthreads();
    const float* xb = x + (size_t)b * 3 * N_;
    float x0 = xb[n], x1 = xb[N_ + n], x2 = xb[2 * N_ + n];
    float d0 = 3.4e38f, d1 = 3.4e38f, d2 = 3.4e38f;
    int i0 = 0, i1 = 0, i2 = 0;
    for (int m = 0; m < M_; ++m) {
        float dx = x0 - nd[m], dy = x1 - nd[64 + m], dz = x2 - nd[128 + m];
        float d = dx * dx + dy * dy + dz * dz;
        if (d < d0)      { d2 = d1; i2 = i1; d1 = d0; i1 = i0; d0 = d; i0 = m; }
        else if (d < d1) { d2 = d1; i2 = i1; d1 = d;  i1 = m; }
        else if (d < d2) { d2 = d;  i2 = m; }
    }
    int base = b * KN_;
    minidx[base + n]            = i0;
    minidx[base + N_ + n]       = i1;
    minidx[base + 2 * N_ + n]   = i2;
}

// ---------------- K1b: deterministic cluster mean (one block per (b,m)) ----
__global__ __launch_bounds__(256) void k_node(const float* __restrict__ x,
                                              const int* __restrict__ minidx,
                                              float* __restrict__ som) {
    int b = blockIdx.x >> 6, m = blockIdx.x & 63;
    int tid = threadIdx.x;
    const int*   mi = minidx + b * KN_;
    const float* xb = x + (size_t)b * 3 * N_;
    float c = 0.f, sx = 0.f, sy = 0.f, sz = 0.f;
    for (int t = tid; t < KN_; t += 256) {
        if (mi[t] == m) {
            int n = t & (N_ - 1);
            c += 1.f; sx += xb[n]; sy += xb[N_ + n]; sz += xb[2 * N_ + n];
        }
    }
    __shared__ float s0[256], s1[256], s2[256], s3[256];
    s0[tid] = c; s1[tid] = sx; s2[tid] = sy; s3[tid] = sz;
    __syncthreads();
    for (int st = 128; st > 0; st >>= 1) {
        if (tid < st) { s0[tid] += s0[tid+st]; s1[tid] += s1[tid+st];
                        s2[tid] += s2[tid+st]; s3[tid] += s3[tid+st]; }
        __syncthreads();
    }
    if (tid == 0) {
        float inv = 1.f / (s0[0] + 1e-5f);
        som[b * 192 + m]        = s1[0] * inv;
        som[b * 192 + 64 + m]   = s2[0] * inv;
        som[b * 192 + 128 + m]  = s3[0] * inv;
    }
}

// ---------------- generic zero kernel ----------------
__global__ __launch_bounds__(256) void k_zero32(u32* p, int n) {
    int i = blockIdx.x * 256 + threadIdx.x;
    if (i < n) p[i] = 0u;
}

// ---------------- weight fp32 row-major -> bf16 [cout][cinPad] -------------
__global__ __launch_bounds__(256) void k_wconv(const float* __restrict__ W,
                                               u16* __restrict__ Wb,
                                               int cin, int cinPad, int n) {
    int gid = blockIdx.x * 256 + threadIdx.x;
    if (gid >= n) return;
    int co = gid / cinPad, ci = gid - co * cinPad;
    Wb[gid] = (ci < cin) ? f2bf(W[(size_t)co * cin + ci]) : (u16)0;
}

// ---------------- K2: build 32-wide bf16 input rows [xdec(3) sn(3) 0..] ----
__global__ __launch_bounds__(256) void k_build_h(const float* __restrict__ x,
                                                 const float* __restrict__ sn,
                                                 const float* __restrict__ som,
                                                 const int* __restrict__ minidx,
                                                 u16* __restrict__ h, int base) {
    int r  = blockIdx.x * 256 + threadIdx.x;          // local chunk row
    int gr = base + r;
    int b = gr / KN_, t = gr - b * KN_, n = t & (N_ - 1);
    int c = minidx[b * KN_ + t];
    const float* xb = x + (size_t)b * 3 * N_;
    const float* sb = sn + (size_t)b * 3 * N_;
    alignas(16) u16 tmp[32];
#pragma unroll
    for (int i = 6; i < 32; ++i) tmp[i] = 0;
    tmp[0] = f2bf(xb[n]          - som[b * 192 + c]);
    tmp[1] = f2bf(xb[N_ + n]     - som[b * 192 + 64 + c]);
    tmp[2] = f2bf(xb[2 * N_ + n] - som[b * 192 + 128 + c]);
    tmp[3] = f2bf(sb[n]);
    tmp[4] = f2bf(sb[N_ + n]);
    tmp[5] = f2bf(sb[2 * N_ + n]);
    uint4*       dst = (uint4*)(h + (size_t)r * 32);
    const uint4* sp  = (const uint4*)tmp;
    dst[0] = sp[0]; dst[1] = sp[1]; dst[2] = sp[2]; dst[3] = sp[3];
}

// ---------------- fused GEMM + bias + BN + ReLU (+ optional segmax) --------
// 128 threads = 4 wave32s; each wave owns a 32row x 64col tile (2 strips).
// Weight tiles double-buffered in LDS, filled by async global->LDS loads.
// A: bf16 [rows x lda], uses cols [0,Kd). W: bf16 [cout][Kd] (K contiguous).
// mode 0: out[row*ldout + colOff + col] = bf16(act)
// mode 1: atomicMax(node_feat_bits[(b*64+m)*384 + col], act)  (fp3 epilogue)
__global__ __launch_bounds__(128) void k_gemm(const u16* __restrict__ A, int lda, int Kd,
                                              const u16* __restrict__ W,
                                              const float* __restrict__ p, int cout,
                                              u16* __restrict__ out, int ldout, int colOff,
                                              int mode, int* __restrict__ nf,
                                              const int* __restrict__ minidx, int chunkBase) {
    __shared__ u16 Bt[2][64 * 32];                    // two 32k x 64n weight tiles
    int tid = threadIdx.x, lane = tid & 31, wv = tid >> 5;
    int rowW = blockIdx.x * 128 + wv * 32;            // 4 waves x 32-row tiles
    int col0 = blockIdx.y * 64;

    v8f acc[2][4] = {};
    int colL = lane & 15;
    int koff = (lane >> 4) * 8;                       // A halves per ISA layout
    int kh   = (lane >> 4) * 8;                       // B halves per ISA layout
    int bc   = tid >> 1;                              // B-stage: col 0..63
    int bkq  = (tid & 1) * 16;                        // B-stage: k 0 / 16
    int nkb  = Kd >> 5;
    const u16* wsrc = W + (size_t)(col0 + bc) * Kd + bkq;

    // prologue: stage tile 0 (2 async b128 per thread per tile)
    {
        u16* dst = &Bt[0][bc * 32 + bkq];
        async_cp16(wsrc, dst);
        async_cp16(wsrc + 8, dst + 8);
    }

    const u16* ap0base = A + (size_t)(rowW + colL) * lda;
    const u16* ap1base = ap0base + (size_t)16 * lda;

    for (int kb = 0; kb < nkb; ++kb) {
        int nxt = kb + 1;
        if (nxt < nkb) {                              // stage next tile, other buffer
            const u16* src = wsrc + (nxt << 5);
            u16* dst = &Bt[nxt & 1][bc * 32 + bkq];
            async_cp16(src, dst);
            async_cp16(src + 8, dst + 8);
            WAIT_ASYNC(2);                            // in-order: tile kb resident
        } else {
            WAIT_ASYNC(0);
        }
        __syncthreads();                              // tile kb visible to all waves

        // A fragments: V0-3 = K[koff..koff+7], V4-7 = K[16+koff..16+koff+7]
        FragB a0, a1;
        const u16* ap0 = ap0base + (kb << 5) + koff;
        const u16* ap1 = ap1base + (kb << 5) + koff;
        a0.q[0] = *(const uint4*)ap0;
        a0.q[1] = *(const uint4*)(ap0 + 16);
        a1.q[0] = *(const uint4*)ap1;
        a1.q[1] = *(const uint4*)(ap1 + 16);

        const u16* bt = Bt[kb & 1];
#pragma unroll
        for (int nt = 0; nt < 4; ++nt) {
            // B fragment: V0-3 = K[kh..kh+7], V4-7 = K[16+kh..16+kh+7] of column
            FragB bfr;
            const u16* bp = &bt[((nt << 4) + colL) * 32 + kh];
            bfr.q[0] = *(const uint4*)bp;
            bfr.q[1] = *(const uint4*)(bp + 16);
            acc[0][nt] = __builtin_amdgcn_wmma_f32_16x16x32_bf16(
                false, a0.v, false, bfr.v, (short)0, acc[0][nt], false, false);
            acc[1][nt] = __builtin_amdgcn_wmma_f32_16x16x32_bf16(
                false, a1.v, false, bfr.v, (short)0, acc[1][nt], false, false);
        }
        __syncthreads();                              // done reading before overwrite
    }

    // epilogue: relu((acc + bias) * gamma * INV_BN + beta)
#pragma unroll
    for (int s = 0; s < 2; ++s) {
        int rbase = rowW + s * 16 + ((lane >> 4) << 3);   // C/D: M = r (+8 hi half)
#pragma unroll
        for (int nt = 0; nt < 4; ++nt) {
            int colG = col0 + (nt << 4) + colL;
            float bb = p[colG];
            float gg = p[cout + colG] * INV_BN_;
            float ee = p[2 * cout + colG];
#pragma unroll
            for (int r = 0; r < 8; ++r) {
                float v = (acc[s][nt][r] + bb) * gg + ee;
                v = v > 0.f ? v : 0.f;
                int row = rbase + r;
                if (mode == 0) {
                    out[(size_t)row * ldout + colOff + colG] = f2bf(v);
                } else {
                    int gr = chunkBase + row;
                    int b  = gr / KN_;
                    int t  = gr - b * KN_;
                    int m  = minidx[b * KN_ + t];
                    atomicMax(nf + ((b << 6) + m) * CF_ + colG, __float_as_int(v));
                }
            }
        }
    }
}

// ---------------- K4a: knn 'avg' centers over SOM nodes --------------------
__global__ __launch_bounds__(256) void k_center(const float* __restrict__ som,
                                                const int* __restrict__ knnI,
                                                float* __restrict__ center) {
    int gid = blockIdx.x * 256 + threadIdx.x;         // B*M
    if (gid >= B_ * M_) return;
    int b = gid >> 6, m = gid & 63;
    float s0 = 0.f, s1 = 0.f, s2 = 0.f;
    for (int s = 0; s < SOMK_; ++s) {
        int j = knnI[gid * SOMK_ + s];
        s0 += som[b * 192 + j];
        s1 += som[b * 192 + 64 + j];
        s2 += som[b * 192 + 128 + j];
    }
    float inv = 1.f / 9.f;
    center[b * 192 + m]       = s0 * inv;
    center[b * 192 + 64 + m]  = s1 * inv;
    center[b * 192 + 128 + m] = s2 * inv;
}

// ---------------- K4b: build knn input g: [co_dec(3) f_nb(384) pad..416] ---
__global__ __launch_bounds__(256) void k_build_g(const float* __restrict__ som,
                                                 const float* __restrict__ center,
                                                 const int* __restrict__ knnI,
                                                 const int* __restrict__ nf,
                                                 u16* __restrict__ g) {
    int gid = blockIdx.x * 256 + threadIdx.x;         // (B*M*SOMK) * 13 chunks
    if (gid >= B_ * M_ * SOMK_ * 13) return;
    int row = gid / 13, cc = gid - row * 13, c0 = cc * 32;
    int bm = row / SOMK_, s = row - bm * SOMK_;
    int b = bm >> 6, m = bm & 63;
    int j = knnI[bm * SOMK_ + s];
    const int* nfr = nf + ((b << 6) + j) * CF_;
    alignas(16) u16 tmp[32];
#pragma unroll
    for (int i = 0; i < 32; ++i) {
        int col = c0 + i;
        float v;
        if (col < 3)        v = som[b * 192 + col * 64 + j] - center[b * 192 + col * 64 + m];
        else if (col < 387) v = __int_as_float(nfr[col - 3]);
        else                v = 0.f;
        tmp[i] = f2bf(v);
    }
    uint4*       dst = (uint4*)(g + (size_t)row * 416 + c0);
    const uint4* sp  = (const uint4*)tmp;
    dst[0] = sp[0]; dst[1] = sp[1]; dst[2] = sp[2]; dst[3] = sp[3];
}

// ---------------- K5: max over SOMK + build z: [center(3) knn(512) pad] ----
__global__ __launch_bounds__(256) void k_build_z(const float* __restrict__ center,
                                                 const u16* __restrict__ k1o,
                                                 u16* __restrict__ z) {
    int gid = blockIdx.x * 256 + threadIdx.x;         // B*M*544
    if (gid >= B_ * M_ * 544) return;
    int bm = gid / 544, col = gid - bm * 544;
    int b = bm >> 6, m = bm & 63;
    float v;
    if (col < 3) {
        v = center[b * 192 + col * 64 + m];
    } else if (col < 515) {
        int c = col - 3;
        const u16* base = k1o + (size_t)bm * SOMK_ * 512 + c;
        float mx = 0.f;
        for (int s = 0; s < SOMK_; ++s) { float t = bf2f(base[s * 512]); mx = t > mx ? t : mx; }
        v = mx;
    } else {
        v = 0.f;
    }
    z[gid] = f2bf(v);
}

// ---------------- K6: global max pool over M nodes -> out (B,1024) f32 -----
__global__ __launch_bounds__(256) void k_final(const u16* __restrict__ f1,
                                               float* __restrict__ out) {
    int gid = blockIdx.x * 256 + threadIdx.x;         // B*1024
    int b = gid >> 10, c = gid & 1023;
    const u16* base = f1 + (size_t)b * M_ * 1024 + c;
    float mx = 0.f;
    for (int m = 0; m < M_; ++m) { float t = bf2f(base[m * 1024]); mx = t > mx ? t : mx; }
    out[gid] = mx;
}

// ============================================================================
extern "C" void kernel_launch(void* const* d_in, const int* in_sizes, int n_in,
                              void* d_out, int out_size, void* d_ws, size_t ws_size,
                              hipStream_t stream) {
    (void)in_sizes; (void)n_in; (void)out_size; (void)ws_size;
    const float* x    = (const float*)d_in[0];
    const float* sn   = (const float*)d_in[1];
    const float* node = (const float*)d_in[2];
    const int*   knnI = (const int*)d_in[3];
    const float* Wf[8]; const float* Pf[8];
    for (int i = 0; i < 8; ++i) { Wf[i] = (const float*)d_in[4 + 2 * i];
                                  Pf[i] = (const float*)d_in[5 + 2 * i]; }
    static const int LCIN[8]  = {6, 64, 128, 320, 387, 512, 515, 768};
    static const int LCINP[8] = {32, 64, 128, 320, 416, 512, 544, 768};
    static const int LCOUT[8] = {64, 128, 256, 384, 512, 512, 768, 1024};

    char* ws = (char*)d_ws;
    size_t off = 0;
    auto take = [&](size_t bytes) -> void* {
        void* ptr = ws + off;
        off += (bytes + 255) & ~(size_t)255;
        return ptr;
    };
    int*   minidx = (int*)take((size_t)ROWS1_ * 4);
    float* som    = (float*)take((size_t)B_ * 3 * M_ * 4);
    float* center = (float*)take((size_t)B_ * 3 * M_ * 4);
    int*   nf     = (int*)take((size_t)B_ * M_ * CF_ * 4);
    u16* Wb[8];
    for (int i = 0; i < 8; ++i) Wb[i] = (u16*)take((size_t)LCOUT[i] * LCINP[i] * 2);
    u16* h    = (u16*)take((size_t)CHR_ * 32 * 2);
    u16* y0y2 = (u16*)take((size_t)CHR_ * 320 * 2);   // fp0 -> cols[0:64), fp2 -> cols[64:320)
    u16* y1   = (u16*)take((size_t)CHR_ * 128 * 2);
    u16* g    = (u16*)take((size_t)B_ * M_ * SOMK_ * 416 * 2);
    u16* kb0  = (u16*)take((size_t)B_ * M_ * SOMK_ * 512 * 2);
    u16* kb1  = (u16*)take((size_t)B_ * M_ * SOMK_ * 512 * 2);
    u16* z    = (u16*)take((size_t)B_ * M_ * 544 * 2);
    u16* f0   = (u16*)take((size_t)B_ * M_ * 768 * 2);
    u16* f1   = (u16*)take((size_t)B_ * M_ * 1024 * 2);

    // weights -> padded bf16
    for (int i = 0; i < 8; ++i) {
        int n = LCOUT[i] * LCINP[i];
        k_wconv<<<(n + 255) / 256, 256, 0, stream>>>(Wf[i], Wb[i], LCIN[i], LCINP[i], n);
    }
    // zero node-feature accumulator (fresh every call; atomicMax target)
    k_zero32<<<(B_ * M_ * CF_ + 255) / 256, 256, 0, stream>>>((u32*)nf, B_ * M_ * CF_);

    // SOM assignment + deterministic cluster means
    k_assign<<<B_ * N_ / 256, 256, 0, stream>>>(x, node, minidx);
    k_node<<<B_ * M_, 256, 0, stream>>>(x, minidx, som);

    // stage 1, chunked: h -> fp0 -> fp1 -> fp2 -> fp3(+fused segment-max)
    for (int ch = 0; ch < NCHUNK_; ++ch) {
        int base = ch * CHR_;
        k_build_h<<<CHR_ / 256, 256, 0, stream>>>(x, sn, som, minidx, h, base);
        k_gemm<<<dim3(CHR_ / 128, 1), 128, 0, stream>>>(h, 32, 32, Wb[0], Pf[0], 64,
                                                        y0y2, 320, 0, 0, nullptr, nullptr, 0);
        k_gemm<<<dim3(CHR_ / 128, 2), 128, 0, stream>>>(y0y2, 320, 64, Wb[1], Pf[1], 128,
                                                        y1, 128, 0, 0, nullptr, nullptr, 0);
        k_gemm<<<dim3(CHR_ / 128, 4), 128, 0, stream>>>(y1, 128, 128, Wb[2], Pf[2], 256,
                                                        y0y2, 320, 64, 0, nullptr, nullptr, 0);
        k_gemm<<<dim3(CHR_ / 128, 6), 128, 0, stream>>>(y0y2, 320, 320, Wb[3], Pf[3], 384,
                                                        nullptr, 0, 0, 1, nf, minidx, base);
    }

    // stage 2: knn module over SOM nodes (4608 rows)
    k_center<<<2, 256, 0, stream>>>(som, knnI, center);
    k_build_g<<<(B_ * M_ * SOMK_ * 13 + 255) / 256, 256, 0, stream>>>(som, center, knnI, nf, g);
    k_gemm<<<dim3(B_ * M_ * SOMK_ / 128, 8), 128, 0, stream>>>(g, 416, 416, Wb[4], Pf[4], 512,
                                                               kb0, 512, 0, 0, nullptr, nullptr, 0);
    k_gemm<<<dim3(B_ * M_ * SOMK_ / 128, 8), 128, 0, stream>>>(kb0, 512, 512, Wb[5], Pf[5], 512,
                                                               kb1, 512, 0, 0, nullptr, nullptr, 0);

    // stage 3: final PointNet (512 rows) + global max pool
    k_build_z<<<(B_ * M_ * 544) / 256, 256, 0, stream>>>(center, kb1, z);
    k_gemm<<<dim3(B_ * M_ / 128, 12), 128, 0, stream>>>(z, 544, 544, Wb[6], Pf[6], 768,
                                                        f0, 768, 0, 0, nullptr, nullptr, 0);
    k_gemm<<<dim3(B_ * M_ / 128, 16), 128, 0, stream>>>(f0, 768, 768, Wb[7], Pf[7], 1024,
                                                        f1, 1024, 0, 0, nullptr, nullptr, 0);
    k_final<<<(B_ * 1024) / 256, 256, 0, stream>>>(f1, (float*)d_out);
}
